// csid_net_90812788506758
// MI455X (gfx1250) — matrix operand order repaired and checked by reading, not compiled, not run on hardware
//
#include <hip/hip_runtime.h>

typedef __attribute__((ext_vector_type(16))) _Float16 v16h;
typedef __attribute__((ext_vector_type(8)))  float    v8f;

#define NDIMS   3
#define HH      256
#define FF      16
#define MM      65536
#define CHUNKS  16                 // K' = 512 = 16 chunks of 32

// ---------------------------------------------------------------------------
// Prep: conditionals c[n,f] = 1 / trapz_t relu(1 + sum_k 2cos*fr - 2sin*fi)
// 48 blocks (one per (n,f)), 128 threads split over the T=100 grid points.
// Hardware v_sin/v_cos take revolutions (angle/2pi) and fract internally.
// ---------------------------------------------------------------------------
__global__ __launch_bounds__(128)
void csid_prep_cond(const float* __restrict__ fr, const float* __restrict__ fi,
                    float* __restrict__ cbuf) {
  __shared__ float red[128];
  const int n = blockIdx.x >> 4;
  const int f = blockIdx.x & 15;
  float acc = 0.0f;
  for (int i = threadIdx.x; i < 100; i += 128) {
    const float t = 0.01f + 0.01f * (float)i;
    float y = 1.0f;                         // k = 0 term (Ar=1, Ai=0)
    float r = t;                            // revolutions for harmonic k
    for (int k = 1; k <= 256; ++k) {
      const int row = 256 - k;              // fr/fi row for harmonic k
      y += 2.0f * __builtin_amdgcn_cosf(r) * fr[(n * HH + row) * FF + f];
      y -= 2.0f * __builtin_amdgcn_sinf(r) * fi[(n * HH + row) * FF + f];
      r += t;
    }
    const float w = (i == 0 || i == 99) ? 0.005f : 0.01f;   // trapezoid weights
    acc += w * fmaxf(y, 0.0f);
  }
  red[threadIdx.x] = acc;
  __syncthreads();
  for (int s = 64; s > 0; s >>= 1) {
    if (threadIdx.x < s) red[threadIdx.x] += red[threadIdx.x + s];
    __syncthreads();
  }
  if (threadIdx.x == 0) cbuf[n * FF + f] = 1.0f / red[0];
}

// ---------------------------------------------------------------------------
// Main: per wave, one 16x16 (samples x factors) tile.
//   proj[n] = 1 + A(16x512 trig, f16) x B(512x16 coeffs, f16)   via WMMA
//   out[m]  = sum_f lam[f] * prod_n relu(proj)*c
// B staged once per block into LDS in exact WMMA B-operand layout.
// A generated in registers in exact WMMA A-operand layout using full-rate
// angle-addition rotations: per-slot (cos,sin) state rotated by the constant
// (cos 32x, sin 32x) each chunk; trans ops only at per-dim init.
// Cos chunk c and sin chunk c+8 share the same phases.
// ---------------------------------------------------------------------------
__global__ __launch_bounds__(256)
void csid_main(const float* __restrict__ x,  const float* __restrict__ fr,
               const float* __restrict__ fi, const float* __restrict__ lam,
               const float* __restrict__ cbuf, float* __restrict__ out) {
  // [n][chunk][lane][16 halves]  -- 49152 bytes
  __shared__ __align__(32) _Float16 sB[NDIMS * CHUNKS * 32 * 16];

  const int tid = threadIdx.x;

  // ---- stage B into LDS (layout: lane holds col N=lane&15, K = 16*(lane>>4)+j)
  for (int idx = tid; idx < NDIMS * CHUNKS * 32 * 16; idx += 256) {
    const int j    = idx & 15;
    const int ln   = (idx >> 4) & 31;
    const int c    = (idx >> 9) & 15;
    const int n    = idx >> 13;
    const int kp   = c * 32 + ((ln >> 4) << 4) + j;   // K' feature index 0..511
    const int f    = ln & 15;
    float v;
    if (kp < 256) v =  2.0f * fr[(n * HH + (255 - kp)) * FF + f];  // cos coeff
    else          v = -2.0f * fi[(n * HH + (511 - kp)) * FF + f];  // sin coeff
    sB[idx] = (_Float16)v;
  }
  __syncthreads();

  const int lane = tid & 31;
  const int wv   = tid >> 5;
  const int g    = lane >> 4;          // half-wave group
  const int f    = lane & 15;          // column / factor index, also A row
  const int tile = blockIdx.x * 8 + wv;
  const int m    = tile * 16 + f;      // sample row generated by this lane (A side)

  const float x0 = x[m * 3 + 0];
  const float x1 = x[m * 3 + 1];
  const float x2 = x[m * 3 + 2];
  const float lamv = lam[f];

  v8f prod = {1.f, 1.f, 1.f, 1.f, 1.f, 1.f, 1.f, 1.f};

  #pragma unroll
  for (int n = 0; n < NDIMS; ++n) {
    const float xv = (n == 0) ? x0 : ((n == 1) ? x1 : x2);
    v8f acc = {1.f, 1.f, 1.f, 1.f, 1.f, 1.f, 1.f, 1.f};   // k=0 term

    // Per-slot (cos,sin) state.  A 16x32 layout: slot j holds
    // K32 = 16*g + j; harmonic for chunk c is k = c*32+K32+1
    // (identical for cos chunk c and sin chunk c+8).
    float cc[16], ss[16];
    #pragma unroll
    for (int j = 0; j < 16; ++j) {
      const int K32 = (g << 4) + j;
      const float r0 = xv * (float)(K32 + 1);     // revolutions
      cc[j] = __builtin_amdgcn_cosf(r0);
      ss[j] = __builtin_amdgcn_sinf(r0);
    }
    const float rs = 32.0f * xv;                  // per-chunk phase step
    const float Cs = __builtin_amdgcn_cosf(rs);
    const float Ss = __builtin_amdgcn_sinf(rs);

    const _Float16* bbase = &sB[(n * CHUNKS) * 32 * 16 + lane * 16];
    for (int c = 0; c < 8; ++c) {
      const v16h bc = *(const v16h*)(bbase + (c    ) * 32 * 16);  // cos coeffs
      const v16h bs = *(const v16h*)(bbase + (c + 8) * 32 * 16);  // sin coeffs
      v16h ac, as;
      #pragma unroll
      for (int j = 0; j < 16; ++j) {
        ac[j] = (_Float16)cc[j];
        as[j] = (_Float16)ss[j];
      }
      acc = __builtin_amdgcn_wmma_f32_16x16x32_f16(false, ac, false, bc,
                                                   (short)0, acc, false, false);
      acc = __builtin_amdgcn_wmma_f32_16x16x32_f16(false, as, false, bs,
                                                   (short)0, acc, false, false);
      if (c != 7) {
        // rotate phases by 32*x revolutions: full-rate FMA pipe, 16 indep chains
        #pragma unroll
        for (int j = 0; j < 16; ++j) {
          const float cn = cc[j] * Cs - ss[j] * Ss;
          const float sn = ss[j] * Cs + cc[j] * Ss;
          cc[j] = cn;
          ss[j] = sn;
        }
      }
    }

    const float cv = cbuf[n * FF + f];
    #pragma unroll
    for (int p = 0; p < 8; ++p)
      prod[p] *= fmaxf(acc[p], 0.0f) * cv;    // relu * conditional, fold product
  }

  // out[m] = sum_f res[m,f] * lam[f] : scale then butterfly over the 16 f-lanes
  #pragma unroll
  for (int p = 0; p < 8; ++p) prod[p] *= lamv;
  #pragma unroll
  for (int off = 1; off < 16; off <<= 1) {
    #pragma unroll
    for (int p = 0; p < 8; ++p)
      prod[p] += __shfl_xor(prod[p], off, 32);
  }

  // D layout: VGPR p holds row 8*g + p; all 16 lanes of a half agree post-reduce
  #pragma unroll
  for (int p = 0; p < 8; ++p)
    if (f == p) out[tile * 16 + g * 8 + p] = prod[p];
}

// ---------------------------------------------------------------------------
extern "C" void kernel_launch(void* const* d_in, const int* in_sizes, int n_in,
                              void* d_out, int out_size, void* d_ws, size_t ws_size,
                              hipStream_t stream) {
  const float* x   = (const float*)d_in[0];
  const float* fr  = (const float*)d_in[1];
  const float* fi  = (const float*)d_in[2];
  const float* lam = (const float*)d_in[3];
  float* cbuf = (float*)d_ws;          // 48 floats of scratch
  float* out  = (float*)d_out;

  csid_prep_cond<<<NDIMS * FF, 128, 0, stream>>>(fr, fi, cbuf);

  const int waves_per_block = 8;                       // 256 threads, wave32
  const int tiles = MM / 16;                           // 4096
  const int grid  = tiles / waves_per_block;           // 512
  csid_main<<<grid, 256, 0, stream>>>(x, fr, fi, lam, cbuf, out);
}
